// MLPRelationGenerator_91199335563522
// MI455X (gfx1250) — matrix operand order repaired and checked by reading, not compiled, not run on hardware
//
#include <hip/hip_runtime.h>

// MLPRelationGenerator for gfx1250 (MI455X, wave32, WMMA f16->f32)
//
// shapes: outs[96][16][512], gs[96][16][512], Wt[512][1024], bt[512],
//         Wp[128][512], bp[128]  -> out[96][16][96][129] (f32 log_softmax)

#define DEPN 96
#define HEADN 96
#define BSZ 16
#define DIM 512
#define RELN 128

typedef __attribute__((ext_vector_type(16))) _Float16 v16h;
typedef __attribute__((ext_vector_type(8)))  float    v8f;
typedef __attribute__((ext_vector_type(2)))  __fp16   h2v;  // cvt_pkrtz result type

union V16 {
  v16h  v;
  h2v   h[8];
  uint4 q[2];
};

struct __align__(4) F4 { float x, y, z, w; };  // 4B-aligned vec store (129-stride rows)

// ---------------------------------------------------------------------------
// Kernel 0: repack weights into WMMA f16 operand layouts (ISA 7.12.2).
// WtB: B-operand layout for the projection GEMMs.
//      chunk index t = ((h*16 + c)*32 + etile)*32 + lane, 16 halves each.
//      lane L: N = e = etile*16 + L%16 ; half k: K = i = 32c + 16*(k/8) + 8*(L/16) + k%8
// WpA: A-operand layout for the pairwise GEMM (A = Wp, M = rel, K = e).
//      chunk index t2 = (c*8 + rt)*32 + lane, 16 halves each.
//      lane L: M = rel = rt*16 + L%16 ;
//      half k: K = 32c + 16*(L/16) + k
// ---------------------------------------------------------------------------
__global__ void mlprel_prep(const float* __restrict__ Wt,
                            const float* __restrict__ Wp,
                            _Float16* __restrict__ WtB,
                            _Float16* __restrict__ WpA) {
  int t = blockIdx.x * blockDim.x + threadIdx.x;
  const int NWT = 2 * 16 * 32 * 32;       // 32768 WtB chunks
  const int NWP = 16 * 8 * 32;            // 4096  WpA chunks
  if (t < NWT) {
    int lane = t & 31, et = (t >> 5) & 31, c = (t >> 10) & 15, h = (t >> 14) & 1;
    int e = et * 16 + (lane & 15), hlf = lane >> 4;
    #pragma unroll
    for (int k = 0; k < 16; ++k) {
      int i = c * 32 + ((k >> 3) << 4) + 8 * hlf + (k & 7);
      WtB[(size_t)t * 16 + k] =
          (_Float16)Wt[(size_t)e * (2 * DIM) + (size_t)h * DIM + i];
    }
  } else if (t < NWT + NWP) {
    int t2 = t - NWT;
    int lane = t2 & 31, rt = (t2 >> 5) & 7, c = t2 >> 8;
    int rel = rt * 16 + (lane & 15), hlf = lane >> 4;
    #pragma unroll
    for (int k = 0; k < 16; ++k) {
      int i = c * 32 + 16 * hlf + k;
      WpA[(size_t)t2 * 16 + k] = (_Float16)Wp[(size_t)rel * DIM + i];
    }
  }
}

// ---------------------------------------------------------------------------
// Kernel 1: projections. One wave computes a 16(row)x64(e) f32 tile.
//   rowTile < 96  : dep_proj[d,b,e] = outs[d,b,:] . Wt[e, :512]   + bt[e]
//   rowTile >= 96 : head_proj[h,b,e] = gs[h,b,:]  . Wt[e, 512:]
// All 4 B tiles for a k-chunk are loaded up front (one clause) so their
// latency overlaps the A-build VALU work; WMMAs then drain them in order.
// ---------------------------------------------------------------------------
__global__ void mlprel_proj(const float* __restrict__ outs,
                            const float* __restrict__ gs,
                            const float* __restrict__ bt,
                            const _Float16* __restrict__ WtB,
                            float* __restrict__ depP,
                            float* __restrict__ headP) {
  int rowTile = blockIdx.x;       // 0..191
  int eg = blockIdx.y;            // 0..7  (64 e-columns each)
  int lane = threadIdx.x;         // wave32
  int b = lane & 15, hlf = lane >> 4;
  bool isDep = rowTile < DEPN;
  int d = isDep ? rowTile : rowTile - DEPN;
  const float* row = (isDep ? outs : gs) + ((size_t)d * BSZ + b) * DIM;
  float* dst = isDep ? depP : headP;
  int h = isDep ? 0 : 1;

  v8f acc[4];
  #pragma unroll
  for (int et = 0; et < 4; ++et) {
    float bv = isDep ? bt[eg * 64 + et * 16 + b] : 0.0f;  // fold bias into C
    #pragma unroll
    for (int v = 0; v < 8; ++v) acc[et][v] = bv;
  }

  for (int c = 0; c < 16; ++c) {
    // issue all B-tile loads first (independent registers -> one clause)
    V16 bm[4];
    #pragma unroll
    for (int et = 0; et < 4; ++et) {
      const uint4* bq = (const uint4*)(WtB +
          ((((size_t)h * 16 + c) * 32 + (eg * 4 + et)) * 32 + lane) * 16);
      bm[et].q[0] = bq[0];
      bm[et].q[1] = bq[1];
    }
    // build A while B loads are in flight
    int i0 = c * 32 + 8 * hlf;
    float4 r0 = *(const float4*)(row + i0);
    float4 r1 = *(const float4*)(row + i0 + 4);
    float4 r2 = *(const float4*)(row + i0 + 16);
    float4 r3 = *(const float4*)(row + i0 + 20);
    V16 a;
    a.h[0] = __builtin_amdgcn_cvt_pkrtz(r0.x, r0.y);
    a.h[1] = __builtin_amdgcn_cvt_pkrtz(r0.z, r0.w);
    a.h[2] = __builtin_amdgcn_cvt_pkrtz(r1.x, r1.y);
    a.h[3] = __builtin_amdgcn_cvt_pkrtz(r1.z, r1.w);
    a.h[4] = __builtin_amdgcn_cvt_pkrtz(r2.x, r2.y);
    a.h[5] = __builtin_amdgcn_cvt_pkrtz(r2.z, r2.w);
    a.h[6] = __builtin_amdgcn_cvt_pkrtz(r3.x, r3.y);
    a.h[7] = __builtin_amdgcn_cvt_pkrtz(r3.z, r3.w);

    #pragma unroll
    for (int et = 0; et < 4; ++et)
      acc[et] = __builtin_amdgcn_wmma_f32_16x16x32_f16(
          false, a.v, false, bm[et].v, (short)0, acc[et], false, false);
  }

  #pragma unroll
  for (int et = 0; et < 4; ++et) {
    int e = eg * 64 + et * 16 + b;
    #pragma unroll
    for (int v = 0; v < 8; ++v) {
      int r = d * BSZ + 8 * hlf + v;          // D layout: lane N=b, VGPR v -> M
      dst[(size_t)r * DIM + e] = acc[et][v];
    }
  }
}

// ---------------------------------------------------------------------------
// Kernel 2: pairwise GEMM + log-softmax. One wave handles (d, 4 heads).
// A = Wp tile (M=rel, preswizzled), B = relu(dep+head) built on the fly in
// the 32x16 f16 B layout (lane L: N = b = L%16, halves k: K = 16*(L/16)+k).
// D lane L: batch = L%16, rel = 16*rt + 8*(L/16) + v. bp folded into C init.
// Per k-chunk: all 8 A tiles issued first (16-load clause), then dep/head
// loads + relu/pack VALU overlap that latency, then 32 WMMAs drain.
// ---------------------------------------------------------------------------
__global__ void mlprel_pair(const float* __restrict__ depP,
                            const float* __restrict__ headP,
                            const _Float16* __restrict__ WpA,
                            const float* __restrict__ bpv,
                            float* __restrict__ out) {
  int d = blockIdx.x;             // 0..95
  int hg = blockIdx.y;            // 0..23 (4 heads each)
  int lane = threadIdx.x;
  int b = lane & 15, hlf = lane >> 4;
  const float* depRow = depP + ((size_t)d * BSZ + b) * DIM;

  v8f acc[4][8];
  #pragma unroll
  for (int rt = 0; rt < 8; ++rt) {
    float4 c0 = *(const float4*)(bpv + rt * 16 + 8 * hlf);
    float4 c1 = *(const float4*)(bpv + rt * 16 + 8 * hlf + 4);
    v8f ini;
    ini[0] = c0.x; ini[1] = c0.y; ini[2] = c0.z; ini[3] = c0.w;
    ini[4] = c1.x; ini[5] = c1.y; ini[6] = c1.z; ini[7] = c1.w;
    #pragma unroll
    for (int p = 0; p < 4; ++p) acc[p][rt] = ini;
  }

  for (int c = 0; c < 16; ++c) {
    // 1) issue all 8 A-tile loads (distinct regs -> single clause, in flight)
    V16 am[8];
    #pragma unroll
    for (int rt = 0; rt < 8; ++rt) {
      const uint4* aq = (const uint4*)(WpA + (((size_t)c * 8 + rt) * 32 + lane) * 16);
      am[rt].q[0] = aq[0];
      am[rt].q[1] = aq[1];
    }

    // 2) dep/head loads + relu + f16 pack (overlaps A-load latency)
    int e0 = c * 32 + 16 * hlf;
    float dv[16];
    {
      float4 a0 = *(const float4*)(depRow + e0);
      float4 a1 = *(const float4*)(depRow + e0 + 4);
      float4 a2 = *(const float4*)(depRow + e0 + 8);
      float4 a3 = *(const float4*)(depRow + e0 + 12);
      dv[0]=a0.x; dv[1]=a0.y; dv[2]=a0.z; dv[3]=a0.w;
      dv[4]=a1.x; dv[5]=a1.y; dv[6]=a1.z; dv[7]=a1.w;
      dv[8]=a2.x; dv[9]=a2.y; dv[10]=a2.z; dv[11]=a2.w;
      dv[12]=a3.x; dv[13]=a3.y; dv[14]=a3.z; dv[15]=a3.w;
    }
    V16 Bm[4];
    #pragma unroll
    for (int p = 0; p < 4; ++p) {
      const float* hRow = headP + ((size_t)(hg * 4 + p) * BSZ + b) * DIM + e0;
      float4 h0 = *(const float4*)(hRow);
      float4 h1 = *(const float4*)(hRow + 4);
      float4 h2 = *(const float4*)(hRow + 8);
      float4 h3 = *(const float4*)(hRow + 12);
      float hv[16];
      hv[0]=h0.x; hv[1]=h0.y; hv[2]=h0.z; hv[3]=h0.w;
      hv[4]=h1.x; hv[5]=h1.y; hv[6]=h1.z; hv[7]=h1.w;
      hv[8]=h2.x; hv[9]=h2.y; hv[10]=h2.z; hv[11]=h2.w;
      hv[12]=h3.x; hv[13]=h3.y; hv[14]=h3.z; hv[15]=h3.w;
      #pragma unroll
      for (int j = 0; j < 8; ++j) {
        float x0 = fmaxf(dv[2*j]   + hv[2*j],   0.0f);
        float x1 = fmaxf(dv[2*j+1] + hv[2*j+1], 0.0f);
        Bm[p].h[j] = __builtin_amdgcn_cvt_pkrtz(x0, x1);
      }
    }

    // 3) 32 WMMAs drain the staged operands
    #pragma unroll
    for (int rt = 0; rt < 8; ++rt)
      #pragma unroll
      for (int p = 0; p < 4; ++p)
        acc[p][rt] = __builtin_amdgcn_wmma_f32_16x16x32_f16(
            false, am[rt].v, false, Bm[p].v, (short)0, acc[p][rt], false, false);
  }

  // log-softmax over [nil, 128 rels]; lanes L and L^16 together cover all 128.
  #pragma unroll
  for (int p = 0; p < 4; ++p) {
    int head = hg * 4 + p;
    size_t base = (((size_t)d * BSZ + b) * HEADN + head) * (RELN + 1);
    float m = -1e30f;
    #pragma unroll
    for (int rt = 0; rt < 8; ++rt)
      #pragma unroll
      for (int v = 0; v < 8; ++v) m = fmaxf(m, acc[p][rt][v]);
    m = fmaxf(m, __shfl_xor(m, 16, 32));
    m = fmaxf(m, 0.0f);                      // include nil score 0
    float s = 0.0f;
    #pragma unroll
    for (int rt = 0; rt < 8; ++rt)
      #pragma unroll
      for (int v = 0; v < 8; ++v) s += __expf(acc[p][rt][v] - m);
    float stot = s + __shfl_xor(s, 16, 32) + __expf(-m);
    float lse = m + __logf(stot);
    if (hlf == 0) out[base] = -lse;          // nil entry
    #pragma unroll
    for (int rt = 0; rt < 8; ++rt) {
      F4* o = (F4*)(out + base + 1 + 16 * rt + 8 * hlf);
      o[0] = F4{acc[p][rt][0] - lse, acc[p][rt][1] - lse,
                acc[p][rt][2] - lse, acc[p][rt][3] - lse};
      o[1] = F4{acc[p][rt][4] - lse, acc[p][rt][5] - lse,
                acc[p][rt][6] - lse, acc[p][rt][7] - lse};
    }
  }
}

// ---------------------------------------------------------------------------
extern "C" void kernel_launch(void* const* d_in, const int* in_sizes, int n_in,
                              void* d_out, int out_size, void* d_ws, size_t ws_size,
                              hipStream_t stream) {
  const float* outs = (const float*)d_in[0];
  const float* gs   = (const float*)d_in[1];
  const float* Wt   = (const float*)d_in[2];
  const float* bt   = (const float*)d_in[3];
  const float* Wp   = (const float*)d_in[4];
  const float* bpv  = (const float*)d_in[5];
  float* out = (float*)d_out;

  // workspace: dep_proj(3MB) | head_proj(3MB) | WtB f16(1MB) | WpA f16(128KB)
  float* depP  = (float*)d_ws;
  float* headP = depP + (size_t)DEPN * BSZ * DIM;
  _Float16* WtB = (_Float16*)(headP + (size_t)HEADN * BSZ * DIM);
  _Float16* WpA = WtB + (size_t)2 * 16 * 32 * 32 * 16;

  const int prepThreads = 2 * 16 * 32 * 32 + 16 * 8 * 32;  // 36864
  mlprel_prep<<<dim3((prepThreads + 255) / 256), dim3(256), 0, stream>>>(
      Wt, Wp, WtB, WpA);
  mlprel_proj<<<dim3(DEPN + HEADN, 8), dim3(32), 0, stream>>>(
      outs, gs, bt, WtB, depP, headP);
  mlprel_pair<<<dim3(DEPN, HEADN / 4), dim3(32), 0, stream>>>(
      depP, headP, WpA, bpv, out);
}
